// ResidualVQ_85401129714121
// MI455X (gfx1250) — matrix operand order repaired and checked by reading, not compiled, not run on hardware
//
#include <hip/hip_runtime.h>

typedef __bf16 bf16_t;
typedef __bf16 v16bf __attribute__((ext_vector_type(16)));
typedef float  v8f   __attribute__((ext_vector_type(8)));

union FragBF { v16bf v; uint4 u[2]; };

#define B_ROWS  65536
#define LATENT  512
#define NEMBED  256
#define NGROUPS 7
#define INDIM   64

// ---------------------------------------------------------------------------
// CDNA5 async global->LDS copy (16 bytes per lane), tracked by ASYNCcnt.
// GVS addressing: saddr = 64-bit SGPR base, vaddr = 32-bit byte offset,
// vdst = LDS byte address.
// ---------------------------------------------------------------------------
__device__ __forceinline__ void async_copy_b128(unsigned lds_addr,
                                                const void* sbase,
                                                unsigned byte_off)
{
    asm volatile("global_load_async_to_lds_b128 %0, %1, %2"
                 :: "v"(lds_addr), "v"(byte_off), "s"(sbase)
                 : "memory");
}
template<int N>
__device__ __forceinline__ void wait_async()
{
    asm volatile("s_wait_asynccnt %0" :: "n"(N) : "memory");
}

// ---------------------------------------------------------------------------
// GEMM: C = act(A @ W + bias). A: MxK bf16 row-major, WT: NxK bf16.
// WG = 256 threads = 8 waves; tile 128(M) x BN(N); K stepped by 32.
// Double-buffered LDS, all staging through async-to-LDS.
// Wave layout: 2(m-half) x 4(n-quarter); each wave 4 m-tiles x (BN/64) n-tiles.
// ---------------------------------------------------------------------------
template<bool RELU, typename OUTT, int BN>
__global__ __launch_bounds__(256)
void gemm_async_kernel(const bf16_t* __restrict__ A, const bf16_t* __restrict__ WT,
                       const float* __restrict__ bias, OUTT* __restrict__ C,
                       int K, int N)
{
    constexpr int BM = 128, BK = 32;
    constexpr int NTW = BN / 64;          // n-tiles per wave (1 or 2)
    constexpr int NISSUE = 2 + BN / 64;   // async instrs per wave per batch

    __shared__ alignas(16) bf16_t As[2][BM][BK];   // 2 x 8 KB
    __shared__ alignas(16) bf16_t Ws[2][BN][BK];   // 2 x (BN/16) KB

    const int tid  = threadIdx.x;
    const int lane = tid & 31;
    const int wid  = tid >> 5;
    const int bm   = blockIdx.x * BM;
    const int bn   = blockIdx.y * BN;
    const int wave_n = (wid & 3) * (BN / 4);
    const int wave_m = (wid >> 2) * 64;

    v8f acc[4][NTW];
#pragma unroll
    for (int t = 0; t < 4; ++t)
#pragma unroll
        for (int j = 0; j < NTW; ++j) acc[t][j] = v8f{};

    // 16-bit A frag: lanes 0-15 -> M=lane, K 0..15; lanes 16-31 -> K 16..31
    const int am = lane & 15;
    const int ak = (lane >> 4) * 16;
    // 16-bit B frag (same contiguous k order as A): lanes 0-15 -> N=lane, K 0..15; lanes 16-31 -> K 16..31
    const int bk = (lane >> 4) * 16;

    auto issue_tiles = [&](int buf, int k0) {
#pragma unroll
        for (int j = 0; j < 2; ++j) {                  // A: 512 x 16B chunks
            const int e = (tid + j * 256) << 3, r = e >> 5, c = e & 31;
            async_copy_b128((unsigned)(uintptr_t)&As[buf][r][c], A,
                            (unsigned)(((size_t)(bm + r) * K + k0 + c) * 2u));
        }
#pragma unroll
        for (int j = 0; j < BN / 64; ++j) {            // W: (BN*4) x 16B chunks
            const int e = (tid + j * 256) << 3, n = e >> 5, c = e & 31;
            async_copy_b128((unsigned)(uintptr_t)&Ws[buf][n][c], WT,
                            (unsigned)(((size_t)(bn + n) * K + k0 + c) * 2u));
        }
    };

    const int KT = K / BK;
    issue_tiles(0, 0);
    for (int kt = 0; kt < KT; ++kt) {
        const int cur = kt & 1;
        if (kt + 1 < KT) {
            issue_tiles(cur ^ 1, (kt + 1) * BK);   // overlap next tile with compute
            wait_async<NISSUE>();                  // current tile complete (in-order)
        } else {
            wait_async<0>();
        }
        __syncthreads();

        FragBF fbs[NTW];
#pragma unroll
        for (int j = 0; j < NTW; ++j) {
            const int n = wave_n + j * 16 + (lane & 15);
            fbs[j].u[0] = *(const uint4*)&Ws[cur][n][bk];
            fbs[j].u[1] = *(const uint4*)&Ws[cur][n][bk + 8];
        }
#pragma unroll
        for (int t = 0; t < 4; ++t) {
            FragBF fa;
            const int row = wave_m + t * 16 + am;
            fa.u[0] = *(const uint4*)&As[cur][row][ak];
            fa.u[1] = *(const uint4*)&As[cur][row][ak + 8];
#pragma unroll
            for (int j = 0; j < NTW; ++j)
                acc[t][j] = __builtin_amdgcn_wmma_f32_16x16x32_bf16(
                    false, fa.v, false, fbs[j].v, (short)0, acc[t][j], false, false);
        }
        __syncthreads();   // all waves done with 'cur' before it is refilled
    }

    // Epilogue: lane holds col n = lane%16; rows M = 8*(lane/16)+v
    const int rsel = (lane >> 4) * 8;
#pragma unroll
    for (int j = 0; j < NTW; ++j) {
        const int cn = bn + wave_n + j * 16 + (lane & 15);
        const float bv = bias[cn];
#pragma unroll
        for (int t = 0; t < 4; ++t)
#pragma unroll
            for (int v = 0; v < 8; ++v) {
                const int cm = bm + wave_m + t * 16 + rsel + v;
                float x = acc[t][j][v] + bv;
                if (RELU) x = fmaxf(x, 0.0f);
                C[(size_t)cm * N + cn] = (OUTT)x;
            }
    }
}

// ---------------------------------------------------------------------------
// One residual-VQ group: scores = r @ E^T via WMMA (bf16), argmin over 256
// codes, then residual -= E[idx]; quantized += E[idx]; SSE into loss_acc.
// WG = 256 threads handles 32 rows. E tile async, R tile fp32->bf16 manual.
// ---------------------------------------------------------------------------
__global__ __launch_bounds__(256)
void vq_group_kernel(const bf16_t* __restrict__ Ebf,   // 256 x 512 bf16
                     const float*  __restrict__ Ef,    // 256 x 512 fp32
                     const float*  __restrict__ Enorm, // 256
                     float* __restrict__ residual,
                     float* __restrict__ quantized,
                     int*   __restrict__ idx_base,     // [b*7 + g]
                     int g,
                     float* __restrict__ loss_acc)
{
    constexpr int VM = 32, BK = 32;
    __shared__ union {
        struct {
            alignas(16) bf16_t Rs[2][VM][BK];      // 2 x 2 KB
            alignas(16) bf16_t Es[2][NEMBED][BK];  // 2 x 16 KB
        } a;
        struct { float dist[VM][NEMBED]; } b;      // 32 KB (after GEMM phase)
    } sm;
    __shared__ int   s_idx[VM];
    __shared__ float s_redv[256];
    __shared__ int   s_redi[256];

    const int tid  = threadIdx.x;
    const int lane = tid & 31;
    const int wid  = tid >> 5;
    const int b0   = blockIdx.x * VM;
    const int wave_m = (wid & 1) * 16;
    const int wave_n = (wid >> 1) * 64;

    v8f acc[4];
#pragma unroll
    for (int t = 0; t < 4; ++t) acc[t] = v8f{};

    const int am = lane & 15;
    const int ak = (lane >> 4) * 16;
    const int bk = (lane >> 4) * 16;

    auto issue_E = [&](int buf, int k0) {          // 4 async instrs per wave
#pragma unroll
        for (int j = 0; j < 4; ++j) {
            const int e = (tid + j * 256) << 3, n = e >> 5, c = e & 31;
            async_copy_b128((unsigned)(uintptr_t)&sm.a.Es[buf][n][c], Ebf,
                            (unsigned)(((size_t)n * LATENT + k0 + c) * 2u));
        }
    };
    auto stage_R = [&](int buf, int k0) {          // fp32 -> bf16, 1 float4/thread
        const int e = tid << 2, r = e >> 5, c = e & 31;
        const float4 f = *(const float4*)(residual + (size_t)(b0 + r) * LATENT + k0 + c);
        sm.a.Rs[buf][r][c + 0] = (bf16_t)f.x; sm.a.Rs[buf][r][c + 1] = (bf16_t)f.y;
        sm.a.Rs[buf][r][c + 2] = (bf16_t)f.z; sm.a.Rs[buf][r][c + 3] = (bf16_t)f.w;
    };

    constexpr int KT = LATENT / BK;                // 16
    issue_E(0, 0);
    stage_R(0, 0);
    for (int kt = 0; kt < KT; ++kt) {
        const int cur = kt & 1;
        if (kt + 1 < KT) {
            issue_E(cur ^ 1, (kt + 1) * BK);
            stage_R(cur ^ 1, (kt + 1) * BK);
            wait_async<4>();
        } else {
            wait_async<0>();
        }
        __syncthreads();

        FragBF fa;
        const int row = wave_m + am;
        fa.u[0] = *(const uint4*)&sm.a.Rs[cur][row][ak];
        fa.u[1] = *(const uint4*)&sm.a.Rs[cur][row][ak + 8];
#pragma unroll
        for (int t = 0; t < 4; ++t) {
            FragBF fb;
            const int n = wave_n + t * 16 + (lane & 15);
            fb.u[0] = *(const uint4*)&sm.a.Es[cur][n][bk];
            fb.u[1] = *(const uint4*)&sm.a.Es[cur][n][bk + 8];
            acc[t] = __builtin_amdgcn_wmma_f32_16x16x32_bf16(
                false, fa.v, false, fb.v, (short)0, acc[t], false, false);
        }
        __syncthreads();
    }

    // dist = ||E||^2 - 2*score (||r||^2 constant per row -> irrelevant to argmin)
    const int rsel = (lane >> 4) * 8;
#pragma unroll
    for (int t = 0; t < 4; ++t) {
        const int n = wave_n + t * 16 + (lane & 15);
        const float en = Enorm[n];
#pragma unroll
        for (int v = 0; v < 8; ++v)
            sm.b.dist[wave_m + rsel + v][n] = en - 2.0f * acc[t][v];
    }
    __syncthreads();

    {   // argmin: 8 threads per row, each scans 32 codes
        const int r = tid >> 3, part = tid & 7;
        float best = 3.4e38f; int bi = 0;
        for (int n = part * 32; n < part * 32 + 32; ++n) {
            const float d = sm.b.dist[r][n];
            if (d < best) { best = d; bi = n; }
        }
        s_redv[tid] = best; s_redi[tid] = bi;
    }
    __syncthreads();
    if ((tid & 7) == 0) {
        const int r = tid >> 3;
        float best = s_redv[tid]; int bi = s_redi[tid];
        for (int j = 1; j < 8; ++j)
            if (s_redv[tid + j] < best) { best = s_redv[tid + j]; bi = s_redi[tid + j]; }
        s_idx[r] = bi;
        idx_base[(size_t)(b0 + r) * NGROUPS + g] = bi;
    }
    __syncthreads();

    // Update residual/quantized (fp32) + commit-loss SSE
    float lsum = 0.0f;
    for (int i = tid; i < VM * LATENT; i += 256) {
        const int r = i >> 9, c = i & 511;
        const float q  = Ef[(size_t)s_idx[r] * LATENT + c];
        const size_t gi = (size_t)(b0 + r) * LATENT + c;
        const float rv = residual[gi];
        const float d  = rv - q;
        lsum += d * d;
        residual[gi]   = d;
        quantized[gi] += q;
    }
    s_redv[tid] = lsum;
    __syncthreads();
    for (int s = 128; s > 0; s >>= 1) {
        if (tid < s) s_redv[tid] += s_redv[tid + s];
        __syncthreads();
    }
    if (tid == 0) atomicAdd(loss_acc, s_redv[0]);
}

// ---------------------------------------------------------------------------
// Small helper kernels
// ---------------------------------------------------------------------------
__global__ void cvt_transpose_kernel(const float* __restrict__ W, bf16_t* __restrict__ WT,
                                     int K, int N)
{
    const int total = K * N;
    for (int i = blockIdx.x * blockDim.x + threadIdx.x; i < total; i += gridDim.x * blockDim.x) {
        const int k = i / N, n = i % N;
        WT[(size_t)n * K + k] = (bf16_t)W[i];
    }
}

__global__ void cvt_kernel(const float* __restrict__ s, bf16_t* __restrict__ d, int n)
{
    for (int i = blockIdx.x * blockDim.x + threadIdx.x; i < n; i += gridDim.x * blockDim.x)
        d[i] = (bf16_t)s[i];
}

__global__ void enorm_kernel(const float* __restrict__ E, float* __restrict__ Enorm)
{
    const int g = blockIdx.x, e = threadIdx.x;
    const float* row = E + ((size_t)g * NEMBED + e) * LATENT;
    float s = 0.0f;
    for (int k = 0; k < LATENT; ++k) s += row[k] * row[k];
    Enorm[g * NEMBED + e] = s;
}

__global__ __launch_bounds__(256)
void recloss_kernel(const float* __restrict__ rec, const float* __restrict__ act,
                    float* __restrict__ acc, int n)
{
    __shared__ float sd[256];
    float s = 0.0f;
    for (int i = blockIdx.x * 256 + threadIdx.x; i < n; i += gridDim.x * 256) {
        const float d = rec[i] - act[i];
        s += d * d;
    }
    sd[threadIdx.x] = s;
    __syncthreads();
    for (int st = 128; st > 0; st >>= 1) {
        if (threadIdx.x < st) sd[threadIdx.x] += sd[threadIdx.x + st];
        __syncthreads();
    }
    if (threadIdx.x == 0) atomicAdd(acc, sd[0]);
}

__global__ void finalize_kernel(const float* __restrict__ loss, float* __restrict__ out)
{
    float t = loss[7] / ((float)B_ROWS * (float)INDIM);
    for (int g = 0; g < NGROUPS; ++g)
        t += 1.25f * loss[g] / ((float)B_ROWS * (float)LATENT);  // (1 + COMMIT_W) * mean
    out[0] = t;
}

// ---------------------------------------------------------------------------
extern "C" void kernel_launch(void* const* d_in, const int* in_sizes, int n_in,
                              void* d_out, int out_size, void* d_ws, size_t ws_size,
                              hipStream_t stream)
{
    (void)in_sizes; (void)n_in; (void)out_size; (void)ws_size;
    const float* actions = (const float*)d_in[0];
    const float* w1  = (const float*)d_in[1];  const float* b1  = (const float*)d_in[2];
    const float* w2  = (const float*)d_in[3];  const float* b2  = (const float*)d_in[4];
    const float* w3  = (const float*)d_in[5];  const float* b3  = (const float*)d_in[6];
    const float* embeds = (const float*)d_in[7];
    const float* dw1 = (const float*)d_in[8];  const float* db1 = (const float*)d_in[9];
    const float* dw2 = (const float*)d_in[10]; const float* db2 = (const float*)d_in[11];
    const float* dw3 = (const float*)d_in[12]; const float* db3 = (const float*)d_in[13];

    char* ws = (char*)d_ws;
    size_t off = 0;
    auto alloc = [&](size_t bytes) -> void* {
        void* p = ws + off;
        off += (bytes + 255) & ~(size_t)255;
        return p;
    };
    float*  residual = (float*)alloc((size_t)B_ROWS * LATENT * 4);    // z, then residual
    float*  quant    = (float*)alloc((size_t)B_ROWS * LATENT * 4);
    bf16_t* act16    = (bf16_t*)alloc((size_t)B_ROWS * INDIM * 2);
    bf16_t* bufA16   = (bf16_t*)alloc((size_t)B_ROWS * LATENT * 2);
    bf16_t* bufB16   = (bf16_t*)alloc((size_t)B_ROWS * LATENT * 2);
    bf16_t* quant16  = (bf16_t*)alloc((size_t)B_ROWS * LATENT * 2);
    bf16_t* w1T   = (bf16_t*)alloc((size_t)LATENT * INDIM * 2);
    bf16_t* w2T   = (bf16_t*)alloc((size_t)LATENT * LATENT * 2);
    bf16_t* w3T   = (bf16_t*)alloc((size_t)LATENT * LATENT * 2);
    bf16_t* dw1T  = (bf16_t*)alloc((size_t)LATENT * LATENT * 2);
    bf16_t* dw2T  = (bf16_t*)alloc((size_t)LATENT * LATENT * 2);
    bf16_t* dw3T  = (bf16_t*)alloc((size_t)INDIM * LATENT * 2);
    bf16_t* Ebf   = (bf16_t*)alloc((size_t)NGROUPS * NEMBED * LATENT * 2);
    float*  Enorm = (float*)alloc((size_t)NGROUPS * NEMBED * 4);
    float*  loss  = (float*)alloc(8 * 4);  // [0..6]=vq SSE per group, [7]=recon SSE

    hipMemsetAsync(quant, 0, (size_t)B_ROWS * LATENT * 4, stream);
    hipMemsetAsync(loss,  0, 8 * 4, stream);

    // bf16 prep: weights transposed to NxK, codebooks, code norms, input activations
    cvt_transpose_kernel<<<128, 256, 0, stream>>>(w1,  w1T,  INDIM,  LATENT);
    cvt_transpose_kernel<<<256, 256, 0, stream>>>(w2,  w2T,  LATENT, LATENT);
    cvt_transpose_kernel<<<256, 256, 0, stream>>>(w3,  w3T,  LATENT, LATENT);
    cvt_transpose_kernel<<<256, 256, 0, stream>>>(dw1, dw1T, LATENT, LATENT);
    cvt_transpose_kernel<<<256, 256, 0, stream>>>(dw2, dw2T, LATENT, LATENT);
    cvt_transpose_kernel<<<128, 256, 0, stream>>>(dw3, dw3T, LATENT, INDIM);
    cvt_kernel<<<512, 256, 0, stream>>>(embeds, Ebf, NGROUPS * NEMBED * LATENT);
    cvt_kernel<<<512, 256, 0, stream>>>(actions, act16, B_ROWS * INDIM);
    enorm_kernel<<<NGROUPS, NEMBED, 0, stream>>>(embeds, Enorm);

    const dim3 blk(256);
    const dim3 g128(B_ROWS / 128, LATENT / 128);   // 128x128 tiles, N=512

    // Encoder: h1 -> bufA16, h2 -> bufB16, z -> residual (fp32)
    gemm_async_kernel<true,  bf16_t, 128><<<g128, blk, 0, stream>>>(act16,  w1T, b1, bufA16,   INDIM,  LATENT);
    gemm_async_kernel<true,  bf16_t, 128><<<g128, blk, 0, stream>>>(bufA16, w2T, b2, bufB16,   LATENT, LATENT);
    gemm_async_kernel<false, float,  128><<<g128, blk, 0, stream>>>(bufB16, w3T, b3, residual, LATENT, LATENT);

    // Residual VQ: 7 sequential groups (true data dependence through residual)
    int* idx_base = (int*)d_out + (size_t)B_ROWS * INDIM;
    for (int g = 0; g < NGROUPS; ++g) {
        vq_group_kernel<<<B_ROWS / 32, blk, 0, stream>>>(
            Ebf + (size_t)g * NEMBED * LATENT,
            embeds + (size_t)g * NEMBED * LATENT,
            Enorm + g * NEMBED,
            residual, quant, idx_base, g, loss + g);
    }

    // Decoder: quant -> bf16, d1 -> bufA16, d2 -> bufB16, rec -> d_out (fp32)
    cvt_kernel<<<512, 256, 0, stream>>>(quant, quant16, B_ROWS * LATENT);
    gemm_async_kernel<true,  bf16_t, 128><<<g128, blk, 0, stream>>>(quant16, dw1T, db1, bufA16, LATENT, LATENT);
    gemm_async_kernel<true,  bf16_t, 128><<<g128, blk, 0, stream>>>(bufA16,  dw2T, db2, bufB16, LATENT, LATENT);
    const dim3 g64(B_ROWS / 128, 1);               // 128x64 tile, N=64
    gemm_async_kernel<false, float,  64 ><<<g64, blk, 0, stream>>>(bufB16, dw3T, db3, (float*)d_out, LATENT, INDIM);

    recloss_kernel<<<1024, blk, 0, stream>>>((const float*)d_out, actions, loss + 7,
                                             B_ROWS * INDIM);
    finalize_kernel<<<1, 1, 0, stream>>>(loss,
        (float*)d_out + (size_t)B_ROWS * INDIM + (size_t)B_ROWS * NGROUPS);
}
